// GraphSAGE_58677843198033
// MI455X (gfx1250) — matrix operand order repaired and checked by reading, not compiled, not run on hardware
//
#include <hip/hip_runtime.h>
#include <stdint.h>
#include <stddef.h>

#define DF 128
#define DE 256
#define DT 18
#define GNUM 128
#define NLAYERS 4

// -------- monotonic float<->uint encoding for atomic max --------
// enc(f) preserves float ordering in unsigned ints. enc(-inf) = 0x007FFFFF,
// used as the "empty segment" sentinel (decoded to 0 per PyG scatter-max).
#define ENC_NEG_INF 0x007FFFFFu

__device__ __forceinline__ unsigned enc_f(float f) {
  unsigned u = __float_as_uint(f);
  return (u & 0x80000000u) ? ~u : (u | 0x80000000u);
}
__device__ __forceinline__ float dec_fix(unsigned e) {
  if (e == ENC_NEG_INF) return 0.0f;  // empty segment -> 0
  return (e & 0x80000000u) ? __uint_as_float(e & 0x7FFFFFFFu)
                           : __uint_as_float(~e);
}

// -------- bf16 hi/lo split (round-to-nearest-even) --------
__device__ __forceinline__ unsigned bf16_rne(float f) {
  unsigned u = __float_as_uint(f);
  return (u + 0x7FFFu + ((u >> 16) & 1u)) >> 16;
}
__device__ __forceinline__ void split2(float f0, float f1,
                                       unsigned& hiP, unsigned& loP) {
  unsigned h0 = bf16_rne(f0), h1 = bf16_rne(f1);
  float r0 = f0 - __uint_as_float(h0 << 16);
  float r1 = f1 - __uint_as_float(h1 << 16);
  unsigned l0 = bf16_rne(r0), l1 = bf16_rne(r1);
  hiP = h0 | (h1 << 16);
  loP = l0 | (l1 << 16);
}

typedef __attribute__((ext_vector_type(16))) __bf16 v16bf;
typedef __attribute__((ext_vector_type(8)))  float  v8f;
typedef __attribute__((ext_vector_type(4)))  unsigned vu4;
union Frag { v16bf v; vu4 q[2]; unsigned u[8]; };

// ----------------------------------------------------------------
// fill
__global__ void fill_u32(unsigned* __restrict__ p, long long n, unsigned v) {
  long long i = (long long)blockIdx.x * blockDim.x + threadIdx.x;
  if (i < n) p[i] = v;
}

// ----------------------------------------------------------------
// edge-parallel scatter-max: agg[dst] = max(agg[dst], h[src]) (encoded)
__global__ void scatter_max(const float* __restrict__ h, int lda,
                            const int* __restrict__ src,
                            const int* __restrict__ dst,
                            int E, int din_log2, unsigned* __restrict__ agg) {
  int tid = threadIdx.x;
  int din = 1 << din_log2;
  int epb = 256 >> din_log2;
  int e = blockIdx.x * epb + (tid >> din_log2);
  int f = tid & (din - 1);
  if (e >= E) return;
  int s = src[e], d = dst[e];
  float v = h[(size_t)s * lda + f];
  atomicMax(&agg[(size_t)d * din + f], enc_f(v));
}

// ----------------------------------------------------------------
// WMMA GEMM: out[M,256] = (sum over passes) A_p @ W_p + bias, opt. ReLU.
// DUAL: two A/W pairs accumulated. dec1: A1 is max-encoded u32 (with -inf fix).
// Split-bf16 (3x v_wmma_f32_16x16x32_bf16 per logical product) for ~f32 accuracy.
// Block = 256 threads = 8 waves; block tile 64(M) x 256(N); wave tile 16 x 128.
// LDS rows padded to 20 dwords (80 B): keeps 16 B alignment for ds_load_b128
// fragment reads and staggers rows across the 64 LDS banks (conflict-free).
#define JPAD 20

template<bool DUAL, bool RELU>
__global__ __launch_bounds__(256) void gemm_wmma(
    const void* __restrict__ A1, int lda1, int dec1,
    const float* __restrict__ A2, int lda2,
    const float* __restrict__ W1, const float* __restrict__ W2,
    const float* __restrict__ bias,
    float* __restrict__ out, int ldo, int M, int K) {
  extern __shared__ unsigned smem[];
  unsigned* Ahi = smem;                 // [64][JPAD]
  unsigned* Alo = Ahi + 64 * JPAD;
  unsigned* Whi = Alo + 64 * JPAD;      // [256][JPAD]
  unsigned* Wlo = Whi + 256 * JPAD;

  const int tid  = threadIdx.x;
  const int lane = tid & 31;
  const int wid  = tid >> 5;
  const int wm   = wid & 3;    // 16-row slice within the 64-row block tile
  const int nh   = wid >> 2;   // 128-col half
  const int lh   = lane >> 4;  // lane half
  const int l16  = lane & 15;
  const int rowBase = blockIdx.x * 64;

  v8f acc[8] = {};

  const int npasses = DUAL ? 2 : 1;
  for (int pass = 0; pass < npasses; ++pass) {
    const void*  A   = (pass == 0) ? A1 : (const void*)A2;
    const int    lda = (pass == 0) ? lda1 : lda2;
    const int    dec = (pass == 0) ? dec1 : 0;
    const float* W   = (pass == 0) ? W1 : W2;

    for (int k0 = 0; k0 < K; k0 += 32) {
      __syncthreads();
      // ---- stage A tile: 64 rows x 16 K-pairs, split to bf16 hi/lo ----
      #pragma unroll
      for (int it = 0; it < 4; ++it) {
        int idx = it * 256 + tid;
        int row = idx >> 4, j = idx & 15;
        int gr = rowBase + row;
        float f0 = 0.0f, f1 = 0.0f;
        if (gr < M) {
          size_t base = (size_t)gr * lda + k0 + 2 * j;
          if (dec) {
            const unsigned* p = (const unsigned*)A + base;
            f0 = dec_fix(p[0]); f1 = dec_fix(p[1]);
          } else {
            const float* p = (const float*)A + base;
            f0 = p[0]; f1 = p[1];
          }
        }
        unsigned hp, lp; split2(f0, f1, hp, lp);
        Ahi[row * JPAD + j] = hp; Alo[row * JPAD + j] = lp;
      }
      // ---- stage W tile: [n][j] with j = K-pair index (transposed) ----
      #pragma unroll
      for (int j = 0; j < 16; ++j) {
        int n = tid;
        float w0 = W[(size_t)(k0 + 2 * j)     * DE + n];
        float w1 = W[(size_t)(k0 + 2 * j + 1) * DE + n];
        unsigned hp, lp; split2(w0, w1, hp, lp);
        Whi[n * JPAD + j] = hp; Wlo[n * JPAD + j] = lp;
      }
      __syncthreads();

      // ---- A fragments (16x32 bf16 layout, ISA 7.12.2), b128 LDS reads ----
      Frag ah, al;
      {
        const unsigned* rh = Ahi + (wm * 16 + l16) * JPAD;
        const unsigned* rl = Alo + (wm * 16 + l16) * JPAD;
        int ab = lh * 4;  // lanes 0-15: j {0-3,8-11}; lanes 16-31: {4-7,12-15}
        ah.q[0] = *(const vu4*)(rh + ab);
        ah.q[1] = *(const vu4*)(rh + ab + 8);
        al.q[0] = *(const vu4*)(rl + ab);
        al.q[1] = *(const vu4*)(rl + ab + 8);
      }
      // ---- per N-tile: B fragments (2x b128 each) + 3 split-bf16 WMMAs ----
      #pragma unroll
      for (int nt = 0; nt < 8; ++nt) {
        int n = nh * 128 + nt * 16 + l16;
        const unsigned* rh = Whi + n * JPAD + lh * 4;  // lanes 0-15: j {0-3,8-11}; 16-31: {4-7,12-15}
        const unsigned* rl = Wlo + n * JPAD + lh * 4;
        Frag bh, bl;
        bh.q[0] = *(const vu4*)(rh);
        bh.q[1] = *(const vu4*)(rh + 8);
        bl.q[0] = *(const vu4*)(rl);
        bl.q[1] = *(const vu4*)(rl + 8);
        acc[nt] = __builtin_amdgcn_wmma_f32_16x16x32_bf16(
            false, ah.v, false, bh.v, (short)0, acc[nt], false, false);
        acc[nt] = __builtin_amdgcn_wmma_f32_16x16x32_bf16(
            false, ah.v, false, bl.v, (short)0, acc[nt], false, false);
        acc[nt] = __builtin_amdgcn_wmma_f32_16x16x32_bf16(
            false, al.v, false, bh.v, (short)0, acc[nt], false, false);
      }
    }
  }

  // ---- epilogue: C layout VGPR r -> M = r + 8*laneHalf, N = lane&15 ----
  #pragma unroll
  for (int nt = 0; nt < 8; ++nt) {
    int n = nh * 128 + nt * 16 + l16;
    float b = bias[n];
    #pragma unroll
    for (int r = 0; r < 8; ++r) {
      int row = rowBase + wm * 16 + lh * 8 + r;
      if (row < M) {
        float v = acc[nt][r] + b;
        if (RELU) v = fmaxf(v, 0.0f);
        out[(size_t)row * ldo + n] = v;
      }
    }
  }
}

// ----------------------------------------------------------------
// global max pool: g_enc[batch[i]][f] = max over nodes (encoded)
__global__ void pool_max(const float* __restrict__ hc,
                         const int* __restrict__ batch,
                         long long total, unsigned* __restrict__ g_enc) {
  long long t = (long long)blockIdx.x * blockDim.x + threadIdx.x;
  if (t >= total) return;
  int i = (int)(t >> 10);      // node  (row pitch 1024)
  int f = (int)(t & 1023);     // feature
  atomicMax(&g_enc[(size_t)batch[i] * 1024 + f], enc_f(hc[t]));
}

// ----------------------------------------------------------------
// last = relu(g @ fc1_W + fc1_b); one block (256 thr) per graph
__global__ void fc1_kernel(const unsigned* __restrict__ g_enc,
                           const float* __restrict__ W,
                           const float* __restrict__ b,
                           float* __restrict__ last) {
  __shared__ float gs[1024];
  int g = blockIdx.x, tid = threadIdx.x;
  for (int k = tid; k < 1024; k += 256) gs[k] = dec_fix(g_enc[g * 1024 + k]);
  __syncthreads();
  float acc = b[tid];
  for (int k = 0; k < 1024; ++k) acc = fmaf(gs[k], W[(size_t)k * DE + tid], acc);
  last[(size_t)g * DE + tid] = fmaxf(acc, 0.0f);
}

// out = last @ fc2_W + fc2_b; logsm = out - logsumexp(out). 1 wave per graph.
__global__ void fc2_kernel(const float* __restrict__ last,
                           const float* __restrict__ W,
                           const float* __restrict__ b,
                           float* __restrict__ logsm,
                           float* __restrict__ raw) {
  __shared__ float ls[256];
  __shared__ float lg[DT];
  __shared__ float s_lse;
  int g = blockIdx.x, tid = threadIdx.x;  // 32 threads
  for (int k = tid; k < 256; k += 32) ls[k] = last[(size_t)g * DE + k];
  __syncthreads();
  if (tid < DT) {
    float acc = b[tid];
    for (int k = 0; k < 256; ++k) acc = fmaf(ls[k], W[(size_t)k * DT + tid], acc);
    lg[tid] = acc;
    raw[g * DT + tid] = acc;
  }
  __syncthreads();
  if (tid == 0) {
    float m = lg[0];
    for (int i = 1; i < DT; ++i) m = fmaxf(m, lg[i]);
    float s = 0.0f;
    for (int i = 0; i < DT; ++i) s += expf(lg[i] - m);
    s_lse = m + logf(s);
  }
  __syncthreads();
  if (tid < DT) logsm[g * DT + tid] = lg[tid] - s_lse;
}

// ----------------------------------------------------------------
extern "C" void kernel_launch(void* const* d_in, const int* in_sizes, int n_in,
                              void* d_out, int out_size, void* d_ws, size_t ws_size,
                              hipStream_t stream) {
  const float* x     = (const float*)d_in[0];
  const int*   ei    = (const int*)d_in[1];
  const int*   batch = (const int*)d_in[2];
  const int Nn = in_sizes[0] / DF;
  const int E  = in_sizes[1] / 2;
  const int* src = ei;
  const int* dst = ei + E;

  const float *Wl[4], *bl[4], *Wr[4], *Wm[4], *bm[4];
  int idx = 3;
  for (int l = 0; l < NLAYERS; ++l) {
    Wl[l] = (const float*)d_in[idx++]; bl[l] = (const float*)d_in[idx++];
    Wr[l] = (const float*)d_in[idx++]; Wm[l] = (const float*)d_in[idx++];
    bm[l] = (const float*)d_in[idx++];
  }
  const float* fc1W = (const float*)d_in[idx++];
  const float* fc1b = (const float*)d_in[idx++];
  const float* fc2W = (const float*)d_in[idx++];
  const float* fc2b = (const float*)d_in[idx++];

  // workspace carve (~308 MB)
  char* ws = (char*)d_ws;
  unsigned* agg   = (unsigned*)ws;  ws += (size_t)Nn * DE * 4;
  float*    tbuf  = (float*)ws;     ws += (size_t)Nn * DE * 4;
  float*    hc    = (float*)ws;     ws += (size_t)Nn * 4 * DE * 4;
  unsigned* g_enc = (unsigned*)ws;  ws += (size_t)GNUM * 4 * DE * 4;

  float* outLogsm = (float*)d_out;
  float* outRaw   = outLogsm + GNUM * DT;
  float* outLast  = outRaw + GNUM * DT;

  const size_t GEMM_LDS = (size_t)(64 * JPAD * 2 + 256 * JPAD * 2) * 4;  // 51200 B
  const int gemmGrid = (Nn + 63) / 64;

  const float* hprev = x;
  int lda = DF, din = DF, din_log2 = 7;
  for (int l = 0; l < NLAYERS; ++l) {
    long long fillN = (long long)Nn * din;
    fill_u32<<<(int)((fillN + 255) / 256), 256, 0, stream>>>(agg, fillN, ENC_NEG_INF);
    int epb = 256 >> din_log2;
    scatter_max<<<(E + epb - 1) / epb, 256, 0, stream>>>(hprev, lda, src, dst, E,
                                                         din_log2, agg);
    // t = dec(agg) @ Wl + hprev @ Wr + bl
    gemm_wmma<true, false><<<gemmGrid, 256, GEMM_LDS, stream>>>(
        agg, din, 1, hprev, lda, Wl[l], Wr[l], bl[l], tbuf, DE, Nn, din);
    // hc[:, l*256:] = relu(t @ Wm + bm)
    gemm_wmma<false, true><<<gemmGrid, 256, GEMM_LDS, stream>>>(
        tbuf, DE, 0, nullptr, 0, Wm[l], nullptr, bm[l],
        hc + (size_t)l * DE, 4 * DE, Nn, DE);
    hprev = hc + (size_t)l * DE;
    lda = 4 * DE; din = DE; din_log2 = 8;
  }

  long long poolFill = (long long)GNUM * 4 * DE;
  fill_u32<<<(int)((poolFill + 255) / 256), 256, 0, stream>>>(g_enc, poolFill,
                                                              ENC_NEG_INF);
  long long poolTot = (long long)Nn * 4 * DE;
  pool_max<<<(int)((poolTot + 255) / 256), 256, 0, stream>>>(hc, batch, poolTot,
                                                             g_enc);
  fc1_kernel<<<GNUM, 256, 0, stream>>>(g_enc, fc1W, fc1b, outLast);
  fc2_kernel<<<GNUM, 32, 0, stream>>>(outLast, fc2W, fc2b, outLogsm, outRaw);
}
